// DecoderRNN_75110388072602
// MI455X (gfx1250) — matrix operand order repaired and checked by reading, not compiled, not run on hardware
//
#include <hip/hip_runtime.h>
#include <hip/hip_bf16.h>

// ---------------------------------------------------------------------------
// Types for CDNA5 WMMA (wave32): V_WMMA_F32_16X16X32_BF16
// ---------------------------------------------------------------------------
typedef __bf16 bf16;
typedef __attribute__((ext_vector_type(8)))  bf16  bf16x8;
typedef __attribute__((ext_vector_type(16))) bf16  bf16x16;
typedef __attribute__((ext_vector_type(8)))  float f32x8;

__device__ __forceinline__ bf16x16 cat16(bf16x8 lo, bf16x8 hi) {
  return __builtin_shufflevector(lo, hi, 0,1,2,3,4,5,6,7,8,9,10,11,12,13,14,15);
}
__device__ __forceinline__ f32x8 wmma_bf16(bf16x16 a, bf16x16 b, f32x8 c) {
  // (neg_a, A, neg_b, B, c_mod, C, reuse_a, reuse_b)
  return __builtin_amdgcn_wmma_f32_16x16x32_bf16(false, a, false, b, (short)0, c,
                                                 false, false);
}

// Problem constants
#define BATCH   128
#define VISN    196
#define VISD    512
#define EMBED   256
#define HID     512
#define VOCAB   10000
#define TSTEPS  20

// ---------------------------------------------------------------------------
// f32 -> bf16 convert (weights / embed table / h0)
// ---------------------------------------------------------------------------
__global__ void k_cvt(const float* __restrict__ s, bf16* __restrict__ d, int n) {
  int i = blockIdx.x * 256 + threadIdx.x;
  if (i < n) d[i] = (bf16)s[i];
}

__global__ void k_bsum(const float* __restrict__ a, const float* __restrict__ b,
                       float* __restrict__ o) {
  int i = blockIdx.x * 256 + threadIdx.x;
  if (i < 4 * HID) o[i] = a[i] + b[i];
}

// ---------------------------------------------------------------------------
// Attention weights: alpha[b,n] = softmax_n(features[b,n,:] . Wv)
// (the h-dependent term is a per-row scalar -> softmax shift-invariant)
// ---------------------------------------------------------------------------
__global__ void k_alpha(const float* __restrict__ features,
                        const float* __restrict__ Wv,
                        float* __restrict__ alpha) {
  int b = blockIdx.x;
  __shared__ float att[VISN];
  __shared__ float red[8];
  int wave = threadIdx.x >> 5, lane = threadIdx.x & 31;
  const float* fb = features + (size_t)b * VISN * VISD;
  for (int n = wave; n < VISN; n += 8) {
    const float* fr = fb + n * VISD;
    float s = 0.f;
    for (int i = lane; i < VISD; i += 32) s += fr[i] * Wv[i];
    for (int o = 16; o; o >>= 1) s += __shfl_xor(s, o, 32);
    if (lane == 0) att[n] = s;
  }
  __syncthreads();
  float m = -3.0e38f;
  if (threadIdx.x < VISN) m = att[threadIdx.x];
  for (int o = 16; o; o >>= 1) m = fmaxf(m, __shfl_xor(m, o, 32));
  if (lane == 0) red[wave] = m;
  __syncthreads();
  if (threadIdx.x == 0) {
    float t = red[0];
    for (int i = 1; i < 8; i++) t = fmaxf(t, red[i]);
    red[0] = t;
  }
  __syncthreads();
  float mx = red[0];
  __syncthreads();
  float e = 0.f;
  if (threadIdx.x < VISN) { e = __expf(att[threadIdx.x] - mx); att[threadIdx.x] = e; }
  float s = e;
  for (int o = 16; o; o >>= 1) s += __shfl_xor(s, o, 32);
  if (lane == 0) red[wave] = s;
  __syncthreads();
  if (threadIdx.x == 0) {
    float t = 0.f;
    for (int i = 0; i < 8; i++) t += red[i];
    red[0] = t;
  }
  __syncthreads();
  float inv = 1.0f / red[0];
  if (threadIdx.x < VISN) alpha[b * VISN + threadIdx.x] = att[threadIdx.x] * inv;
}

// ---------------------------------------------------------------------------
// ctx[b,d] = sum_n alpha[b,n]*features[b,n,d] ; fbar[b,d] = mean_n features
// ---------------------------------------------------------------------------
__global__ void k_ctx_fbar(const float* __restrict__ features,
                           const float* __restrict__ alpha,
                           bf16* __restrict__ ctxB, bf16* __restrict__ fbarB) {
  int b = blockIdx.x;
  __shared__ float al[VISN];
  for (int i = threadIdx.x; i < VISN; i += 256) al[i] = alpha[b * VISN + i];
  __syncthreads();
  const float* fb = features + (size_t)b * VISN * VISD;
  for (int d = threadIdx.x; d < VISD; d += 256) {
    float sc = 0.f, sf = 0.f;
    for (int n = 0; n < VISN; n++) {
      float v = fb[n * VISD + d];
      sc += al[n] * v;
      sf += v;
    }
    ctxB[b * VISD + d]  = (bf16)sc;
    fbarB[b * VISD + d] = (bf16)(sf * (1.0f / (float)VISN));
  }
}

// ---------------------------------------------------------------------------
// Generic WMMA GEMM: Cout[M,N] = A[M,K](bf16) * W[N,K](bf16)^T (+ bias)
// Each wave computes a 16x64 strip (4 N-tiles) sharing ONE A fragment per
// k-step: 2 A loads + 8 B loads per 4 wmma.  The tail strip is handled by
// CLAMPING B-row indices (branchless, loads stay in-bounds) so the K-loop
// has NO control flow; only the final stores are per-tile guarded.
// ---------------------------------------------------------------------------
__global__ void wmma_gemm(const bf16* __restrict__ A, int lda,
                          const bf16* __restrict__ W, int ldw,
                          const float* __restrict__ bias,
                          float* __restrict__ Cout, int ldc,
                          int M, int N, int K) {
  const int tilesN  = N >> 4;
  const int stripsN = (tilesN + 3) >> 2;
  const int wid = blockIdx.x * (blockDim.x >> 5) + (threadIdx.x >> 5);
  const int mt = wid / stripsN, st = wid % stripsN;
  if (mt * 16 >= M) return;                       // wave-uniform exit
  const int nt0 = st << 2;
  const int lane = threadIdx.x & 31;
  const int l16  = lane & 15;
  const int half = lane >> 4;
  const int m0 = mt << 4;

  // clamped B-row (= output column) index per tile: always in-bounds
  int col0 = (nt0 + 0) * 16 + l16; col0 = col0 < N ? col0 : N - 1;
  int col1 = (nt0 + 1) * 16 + l16; col1 = col1 < N ? col1 : N - 1;
  int col2 = (nt0 + 2) * 16 + l16; col2 = col2 < N ? col2 : N - 1;
  int col3 = (nt0 + 3) * 16 + l16; col3 = col3 < N ? col3 : N - 1;

  f32x8 acc[4];
  {
    float b0 = bias ? bias[col0] : 0.f;
    float b1 = bias ? bias[col1] : 0.f;
    float b2 = bias ? bias[col2] : 0.f;
    float b3 = bias ? bias[col3] : 0.f;
    #pragma unroll
    for (int i = 0; i < 8; i++) {
      acc[0][i] = b0; acc[1][i] = b1; acc[2][i] = b2; acc[3][i] = b3;
    }
  }

  const bf16* arow = A + (size_t)(m0 + l16) * lda + half * 16;
  const bf16* w0 = W + (size_t)col0 * ldw + half * 16;
  const bf16* w1 = W + (size_t)col1 * ldw + half * 16;
  const bf16* w2 = W + (size_t)col2 * ldw + half * 16;
  const bf16* w3 = W + (size_t)col3 * ldw + half * 16;

  for (int k = 0; k < K; k += 32) {
    bf16x16 av = cat16(*(const bf16x8*)(arow + k), *(const bf16x8*)(arow + k + 8));
    acc[0] = wmma_bf16(av, cat16(*(const bf16x8*)(w0 + k), *(const bf16x8*)(w0 + k + 8)), acc[0]);
    acc[1] = wmma_bf16(av, cat16(*(const bf16x8*)(w1 + k), *(const bf16x8*)(w1 + k + 8)), acc[1]);
    acc[2] = wmma_bf16(av, cat16(*(const bf16x8*)(w2 + k), *(const bf16x8*)(w2 + k + 8)), acc[2]);
    acc[3] = wmma_bf16(av, cat16(*(const bf16x8*)(w3 + k), *(const bf16x8*)(w3 + k + 8)), acc[3]);
  }

  #pragma unroll
  for (int j = 0; j < 4; j++) {
    if (nt0 + j < tilesN) {                       // tail guard: outside hot loop
      const int n0 = (nt0 + j) << 4;
      #pragma unroll
      for (int i = 0; i < 8; i++)
        Cout[(size_t)(m0 + half * 8 + i) * ldc + n0 + l16] = acc[j][i];
    }
  }
}

// ---------------------------------------------------------------------------
// Per-step recurrent GEMM, fused embedding gather:
//   gates = gates_const + emb_t @ W_ih[:,512:768]^T + h @ W_hh^T
// M=128, N=2048 -> 8 x 32 strips of 16x64; 32 blocks x 8 waves; no tails.
// ---------------------------------------------------------------------------
__global__ void k_gates_step(const bf16* __restrict__ Emb,
                             const int* __restrict__ cap, int t,
                             const bf16* __restrict__ hbf,
                             const bf16* __restrict__ Wih,   // [2048,768] bf16
                             const bf16* __restrict__ Whh,   // [2048,512] bf16
                             const float* __restrict__ gconst,
                             float* __restrict__ gates) {
  const int wid  = blockIdx.x * 8 + (threadIdx.x >> 5);      // 256 waves
  const int mt = wid >> 5, st = wid & 31;
  const int lane = threadIdx.x & 31;
  const int l16  = lane & 15;
  const int half = lane >> 4;
  const int m0 = mt << 4, n00 = st << 6;

  f32x8 acc[4];
  #pragma unroll
  for (int j = 0; j < 4; j++)
    #pragma unroll
    for (int i = 0; i < 8; i++)
      acc[j][i] = gconst[(size_t)(m0 + half * 8 + i) * (4 * HID) + n00 + j * 16 + l16];

  const int b    = m0 + l16;
  const int widx = cap[b * TSTEPS + t];

  // phase 1: embedding part, K = 256
  {
    const bf16* arow = Emb + (size_t)widx * EMBED + half * 16;
    const bf16* w0 = Wih + (size_t)(n00 +  0 + l16) * (VISD + EMBED) + VISD + half * 16;
    const bf16* w1 = Wih + (size_t)(n00 + 16 + l16) * (VISD + EMBED) + VISD + half * 16;
    const bf16* w2 = Wih + (size_t)(n00 + 32 + l16) * (VISD + EMBED) + VISD + half * 16;
    const bf16* w3 = Wih + (size_t)(n00 + 48 + l16) * (VISD + EMBED) + VISD + half * 16;
    for (int k = 0; k < EMBED; k += 32) {
      bf16x16 av = cat16(*(const bf16x8*)(arow + k), *(const bf16x8*)(arow + k + 8));
      acc[0] = wmma_bf16(av, cat16(*(const bf16x8*)(w0 + k), *(const bf16x8*)(w0 + k + 8)), acc[0]);
      acc[1] = wmma_bf16(av, cat16(*(const bf16x8*)(w1 + k), *(const bf16x8*)(w1 + k + 8)), acc[1]);
      acc[2] = wmma_bf16(av, cat16(*(const bf16x8*)(w2 + k), *(const bf16x8*)(w2 + k + 8)), acc[2]);
      acc[3] = wmma_bf16(av, cat16(*(const bf16x8*)(w3 + k), *(const bf16x8*)(w3 + k + 8)), acc[3]);
    }
  }
  // phase 2: recurrent part, K = 512
  {
    const bf16* arow = hbf + (size_t)b * HID + half * 16;
    const bf16* w0 = Whh + (size_t)(n00 +  0 + l16) * HID + half * 16;
    const bf16* w1 = Whh + (size_t)(n00 + 16 + l16) * HID + half * 16;
    const bf16* w2 = Whh + (size_t)(n00 + 32 + l16) * HID + half * 16;
    const bf16* w3 = Whh + (size_t)(n00 + 48 + l16) * HID + half * 16;
    for (int k = 0; k < HID; k += 32) {
      bf16x16 av = cat16(*(const bf16x8*)(arow + k), *(const bf16x8*)(arow + k + 8));
      acc[0] = wmma_bf16(av, cat16(*(const bf16x8*)(w0 + k), *(const bf16x8*)(w0 + k + 8)), acc[0]);
      acc[1] = wmma_bf16(av, cat16(*(const bf16x8*)(w1 + k), *(const bf16x8*)(w1 + k + 8)), acc[1]);
      acc[2] = wmma_bf16(av, cat16(*(const bf16x8*)(w2 + k), *(const bf16x8*)(w2 + k + 8)), acc[2]);
      acc[3] = wmma_bf16(av, cat16(*(const bf16x8*)(w3 + k), *(const bf16x8*)(w3 + k + 8)), acc[3]);
    }
  }

  #pragma unroll
  for (int j = 0; j < 4; j++)
    #pragma unroll
    for (int i = 0; i < 8; i++)
      gates[(size_t)(m0 + half * 8 + i) * (4 * HID) + n00 + j * 16 + l16] = acc[j][i];
}

// ---------------------------------------------------------------------------
// LSTM elementwise (torch gate order i,f,g,o); c in f32, h written as bf16
// ---------------------------------------------------------------------------
__global__ void k_lstm(const float* __restrict__ gates,
                       float* __restrict__ c, bf16* __restrict__ hbf) {
  int idx = blockIdx.x * 256 + threadIdx.x;        // 128*512 total
  int b = idx >> 9, j = idx & 511;
  const float* g = gates + (size_t)b * (4 * HID);
  float gi = g[j], gf = g[HID + j], gg = g[2 * HID + j], go = g[3 * HID + j];
  float si = 1.f / (1.f + __expf(-gi));
  float sf = 1.f / (1.f + __expf(-gf));
  float so = 1.f / (1.f + __expf(-go));
  float cn = sf * c[idx] + si * tanhf(gg);
  c[idx] = cn;
  hbf[idx] = (bf16)(so * tanhf(cn));
}

// ---------------------------------------------------------------------------
// Final: rows of d_out[0 .. T*B*V) currently hold logits; overwrite with
// log_softmax in place, write softmax to second half. One block per row.
// ---------------------------------------------------------------------------
__global__ void k_softmax(float* __restrict__ out) {
  const size_t TOT = (size_t)TSTEPS * BATCH * VOCAB;
  int r = blockIdx.x;
  float* row  = out + (size_t)r * VOCAB;
  float* srow = row + TOT;
  __shared__ float sh[8];
  int lane = threadIdx.x & 31, wv = threadIdx.x >> 5;

  float m = -3.0e38f;
  for (int i = threadIdx.x; i < VOCAB; i += 256) m = fmaxf(m, row[i]);
  for (int o = 16; o; o >>= 1) m = fmaxf(m, __shfl_xor(m, o, 32));
  if (lane == 0) sh[wv] = m;
  __syncthreads();
  if (threadIdx.x == 0) {
    float t = sh[0];
    for (int i = 1; i < 8; i++) t = fmaxf(t, sh[i]);
    sh[0] = t;
  }
  __syncthreads();
  float mx = sh[0];
  __syncthreads();

  float s = 0.f;
  for (int i = threadIdx.x; i < VOCAB; i += 256) s += __expf(row[i] - mx);
  for (int o = 16; o; o >>= 1) s += __shfl_xor(s, o, 32);
  if (lane == 0) sh[wv] = s;
  __syncthreads();
  if (threadIdx.x == 0) {
    float t = 0.f;
    for (int i = 0; i < 8; i++) t += sh[i];
    sh[0] = t;
  }
  __syncthreads();
  float sum = sh[0];
  float lg = __logf(sum), inv = 1.0f / sum;
  for (int i = threadIdx.x; i < VOCAB; i += 256) {
    float x = row[i] - mx;
    srow[i] = __expf(x) * inv;
    row[i]  = x - lg;
  }
}

// ---------------------------------------------------------------------------
// Workspace layout (bytes, all 256B aligned)
// ---------------------------------------------------------------------------
static const size_t OFF_WIH    = 0;                       // 2048*768  bf16
static const size_t OFF_WINH   = OFF_WIH    + 3145728;    // 512*512   bf16
static const size_t OFF_WINC   = OFF_WINH   + 524288;     // 512*512   bf16
static const size_t OFF_WHH    = OFF_WINC   + 524288;     // 2048*512  bf16
static const size_t OFF_WO     = OFF_WHH    + 2097152;    // 10000*512 bf16
static const size_t OFF_EMB    = OFF_WO     + 10240000;   // 10000*256 bf16
static const size_t OFF_ALPHA  = OFF_EMB    + 5120000;    // 128*196   f32
static const size_t OFF_CTXB   = OFF_ALPHA  + 100352;     // 128*512   bf16
static const size_t OFF_FBARB  = OFF_CTXB   + 131072;     // 128*512   bf16
static const size_t OFF_HBF    = OFF_FBARB  + 131072;     // 128*512   bf16
static const size_t OFF_HF     = OFF_HBF    + 131072;     // 128*512   f32
static const size_t OFF_C      = OFF_HF     + 262144;     // 128*512   f32
static const size_t OFF_GATES  = OFF_C      + 262144;     // 128*2048  f32
static const size_t OFF_GCONST = OFF_GATES  + 1048576;    // 128*2048  f32
static const size_t OFF_BSUM   = OFF_GCONST + 1048576;    // 2048      f32
static const size_t WS_NEEDED  = OFF_BSUM   + 8192;

extern "C" void kernel_launch(void* const* d_in, const int* in_sizes, int n_in,
                              void* d_out, int out_size, void* d_ws, size_t ws_size,
                              hipStream_t stream) {
  (void)in_sizes; (void)n_in; (void)out_size;
  if (ws_size < WS_NEEDED) return;

  const float* features = (const float*)d_in[0];
  const int*   captions = (const int*)d_in[1];
  const float* W_init_h = (const float*)d_in[2];
  const float* W_init_c = (const float*)d_in[3];
  const float* Wv       = (const float*)d_in[4];
  // d_in[5]=bv, d_in[6]=Wa, d_in[7]=ba cancel out of softmax (shift-invariance)
  const float* embed    = (const float*)d_in[8];
  const float* W_ih     = (const float*)d_in[9];
  const float* W_hh     = (const float*)d_in[10];
  const float* b_ih     = (const float*)d_in[11];
  const float* b_hh     = (const float*)d_in[12];
  const float* Wo       = (const float*)d_in[13];
  const float* bo       = (const float*)d_in[14];

  float* out = (float*)d_out;
  char*  ws  = (char*)d_ws;

  bf16*  WihB   = (bf16*)(ws + OFF_WIH);
  bf16*  WinHB  = (bf16*)(ws + OFF_WINH);
  bf16*  WinCB  = (bf16*)(ws + OFF_WINC);
  bf16*  WhhB   = (bf16*)(ws + OFF_WHH);
  bf16*  WoB    = (bf16*)(ws + OFF_WO);
  bf16*  EmbB   = (bf16*)(ws + OFF_EMB);
  float* alpha  = (float*)(ws + OFF_ALPHA);
  bf16*  ctxB   = (bf16*)(ws + OFF_CTXB);
  bf16*  fbarB  = (bf16*)(ws + OFF_FBARB);
  bf16*  hbf    = (bf16*)(ws + OFF_HBF);
  float* hf     = (float*)(ws + OFF_HF);
  float* cbuf   = (float*)(ws + OFF_C);
  float* gates  = (float*)(ws + OFF_GATES);
  float* gconst = (float*)(ws + OFF_GCONST);
  float* bsum   = (float*)(ws + OFF_BSUM);

  // ---- one-time weight conversion to bf16 (L2-resident afterwards) ----
  k_cvt<<<(HID * VISD + 255) / 256, 256, 0, stream>>>(W_init_h, WinHB, HID * VISD);
  k_cvt<<<(HID * VISD + 255) / 256, 256, 0, stream>>>(W_init_c, WinCB, HID * VISD);
  k_cvt<<<(4 * HID * (VISD + EMBED) + 255) / 256, 256, 0, stream>>>(
      W_ih, WihB, 4 * HID * (VISD + EMBED));
  k_cvt<<<(4 * HID * HID + 255) / 256, 256, 0, stream>>>(W_hh, WhhB, 4 * HID * HID);
  k_cvt<<<(VOCAB * HID + 255) / 256, 256, 0, stream>>>(Wo, WoB, VOCAB * HID);
  k_cvt<<<(VOCAB * EMBED + 255) / 256, 256, 0, stream>>>(embed, EmbB, VOCAB * EMBED);
  k_bsum<<<(4 * HID + 255) / 256, 256, 0, stream>>>(b_ih, b_hh, bsum);

  // ---- loop-invariant attention / context / initial state ----
  k_alpha<<<BATCH, 256, 0, stream>>>(features, Wv, alpha);
  k_ctx_fbar<<<BATCH, 256, 0, stream>>>(features, alpha, ctxB, fbarB);

  // h0 = fbar @ W_init_h^T ; c0 = fbar @ W_init_c^T (128x512, K=512):
  // 8 m-tiles x 8 strips = 64 waves -> 8 blocks
  wmma_gemm<<<8, 256, 0, stream>>>(fbarB, VISD, WinHB, VISD, nullptr,
                                   hf, HID, BATCH, HID, VISD);
  k_cvt<<<(BATCH * HID + 255) / 256, 256, 0, stream>>>(hf, hbf, BATCH * HID);
  wmma_gemm<<<8, 256, 0, stream>>>(fbarB, VISD, WinCB, VISD, nullptr,
                                   cbuf, HID, BATCH, HID, VISD);

  // gates_const = ctx @ W_ih[:, :512]^T + (b_ih + b_hh)  (128x2048, K=512):
  // 8 x 32 strips = 256 waves -> 32 blocks
  wmma_gemm<<<32, 256, 0, stream>>>(ctxB, VISD, WihB, VISD + EMBED, bsum,
                                    gconst, 4 * HID, BATCH, 4 * HID, VISD);

  // ---- sequential decode ----
  for (int t = 0; t < TSTEPS; t++) {
    k_gates_step<<<32, 256, 0, stream>>>(EmbB, captions, t, hbf, WihB, WhhB,
                                         gconst, gates);
    k_lstm<<<(BATCH * HID) / 256, 256, 0, stream>>>(gates, cbuf, hbf);
    // logits_t = h @ Wo^T + bo -> time-major rows of d_out (128x10000, K=512):
    // 8 m-tiles x 157 strips = 1256 waves -> 157 blocks
    wmma_gemm<<<157, 256, 0, stream>>>(hbf, HID, WoB, HID, bo,
                                       out + (size_t)t * BATCH * VOCAB, VOCAB,
                                       BATCH, VOCAB, HID);
  }

  // ---- log_softmax in place + softmax into second half of d_out ----
  k_softmax<<<TSTEPS * BATCH, 256, 0, stream>>>(out);
}
